// AutoregressiveLAMDecoder_84232898609494
// MI455X (gfx1250) — compile-verified, matrix operand order checked by reading, not run on hardware
//
// MI455X (gfx1250) implementation of the autoregressive LAM decoder.
// All large GEMMs run through v_wmma_f32_16x16x32_bf16 with fp32 accumulation,
// using a register+LDS double-buffered pipeline (one barrier per k-step,
// global loads for tile k+1 overlap WMMAs on tile k). LN/residual chains stay
// fp32. Cross-attention is simplified exactly (softmax over a single key == 1),
// removing ca_wq/ca_wk/ln2 entirely.
#include <hip/hip_runtime.h>
#include <cstddef>
#include <cstdint>

typedef __bf16 bf16;
typedef __attribute__((ext_vector_type(16))) __bf16 v16bf;
typedef __attribute__((ext_vector_type(4)))  __bf16 bf16x4;
typedef __attribute__((ext_vector_type(8)))  float  v8f;
typedef __attribute__((ext_vector_type(4)))  float  f32x4;
typedef __attribute__((ext_vector_type(4)))  unsigned int u32x4;

#define B_SZ   16384
#define D_SZ   1024
#define H_SZ   8
#define NT_SZ  4
#define NC_SZ  16
#define DFF_SZ 2048
#define M_DEC  (B_SZ * NT_SZ)   // 65536 decoder rows

// ---------------------------------------------------------------- fp32 -> bf16
__global__ void __launch_bounds__(256)
cvt_bf16_kernel(const float* __restrict__ in, bf16* __restrict__ out, int n)
{
    int i = blockIdx.x * 256 + threadIdx.x;
    if (i < n) out[i] = (bf16)in[i];
}

// ------------------------------------------------- LayerNorm(row of 1024) -> bf16
// one wave (32 lanes) per row; 8 rows per 256-thread block
__global__ void __launch_bounds__(256)
ln_bf16_kernel(const float* __restrict__ X, const float* __restrict__ gamma,
               const float* __restrict__ beta, bf16* __restrict__ Y)
{
    const int row  = blockIdx.x * 8 + (threadIdx.x >> 5);
    const int lane = threadIdx.x & 31;
    const float* x = X + (size_t)row * D_SZ;
    float s = 0.f, s2 = 0.f;
    f32x4 vv[8];
#pragma unroll
    for (int i = 0; i < 8; ++i) {
        f32x4 v = ((const f32x4*)x)[lane + 32 * i];
        vv[i] = v;
        s  += v.x + v.y + v.z + v.w;
        s2 += v.x * v.x + v.y * v.y + v.z * v.z + v.w * v.w;
    }
#pragma unroll
    for (int off = 16; off > 0; off >>= 1) {
        s  += __shfl_xor(s,  off, 32);
        s2 += __shfl_xor(s2, off, 32);
    }
    const float m   = s  * (1.0f / D_SZ);
    const float var = s2 * (1.0f / D_SZ) - m * m;
    const float inv = rsqrtf(var + 1e-5f);
    bf16* y = Y + (size_t)row * D_SZ;
#pragma unroll
    for (int i = 0; i < 8; ++i) {
        const int gi = lane + 32 * i;
        f32x4 g  = ((const f32x4*)gamma)[gi];
        f32x4 bt = ((const f32x4*)beta)[gi];
        f32x4 v  = vv[i];
        bf16x4 o;
        o.x = (bf16)((v.x - m) * inv * g.x + bt.x);
        o.y = (bf16)((v.y - m) * inv * g.y + bt.y);
        o.z = (bf16)((v.z - m) * inv * g.z + bt.z);
        o.w = (bf16)((v.w - m) * inv * g.w + bt.w);
        ((bf16x4*)y)[gi] = o;
    }
}

// ----------------------------------------------------------- tiled WMMA GEMM
// C[M,N] = epi( A[M,K](bf16) @ W[N,K]^T(bf16) + bias[N] (+ resid[M,N]) )
// block: 256 thr (8 waves), 128x128 tile, K-step 32, double-buffered LDS:
//   iter k: issue global loads for tile k+1 -> regs,
//           ds_load frags from stage k&1, 8 WMMAs,
//           ds_store regs -> stage (k+1)&1, single barrier.
// wave grid 4(M)x2(N): each wave owns 32x64 => 2x4 16x16 accumulators.
// EPI: 0=none, 1=gelu(exact), 2=relu
template <int EPI, bool RESID, bool OUTF, bool OUTB>
__global__ void __launch_bounds__(256)
gemm_bf16_wmma(const bf16* __restrict__ A, const bf16* __restrict__ W,
               const float* __restrict__ bias, const float* __restrict__ resid,
               float* __restrict__ outF, bf16* __restrict__ outB,
               int M, int N, int K)
{
    __shared__ bf16 ldsA[2][128 * 40];   // 32 K-halfs + 8 pad (80B row stride)
    __shared__ bf16 ldsW[2][128 * 40];   // 2 stages x 10240 B x 2 operands = 40KB

    const int tid  = threadIdx.x;
    const int wave = tid >> 5;
    const int lane = tid & 31;
    const int g    = lane >> 4;   // lane half-group (WMMA K-chunk selector)
    const int o    = lane & 15;   // M index (A frag) / N index (B frag)
    const int bm = blockIdx.x * 128;
    const int bn = blockIdx.y * 128;
    const int wm = (wave & 3) * 32;
    const int wn = (wave >> 2) * 64;

    v8f acc[2][4];
#pragma unroll
    for (int mt = 0; mt < 2; ++mt)
#pragma unroll
        for (int nt = 0; nt < 4; ++nt)
#pragma unroll
            for (int r = 0; r < 8; ++r) acc[mt][nt][r] = 0.f;

    // each thread stages 2 16-byte chunks per operand per tile
    u32x4 ra[2], rw[2];
    auto load_tile = [&](int kk) {
#pragma unroll
        for (int it = 0; it < 2; ++it) {
            const int c = tid + it * 256, row = c >> 2, cc = c & 3;
            ra[it] = *(const u32x4*)(A + (size_t)(bm + row) * K + kk + cc * 8);
            rw[it] = *(const u32x4*)(W + (size_t)(bn + row) * K + kk + cc * 8);
        }
    };
    auto store_tile = [&](int s) {
#pragma unroll
        for (int it = 0; it < 2; ++it) {
            const int c = tid + it * 256, row = c >> 2, cc = c & 3;
            *(u32x4*)&ldsA[s][row * 40 + cc * 8] = ra[it];
            *(u32x4*)&ldsW[s][row * 40 + cc * 8] = rw[it];
        }
    };

    load_tile(0);
    store_tile(0);
    __syncthreads();

    const int nk = K >> 5;
    for (int k = 0; k < nk; ++k) {
        if (k + 1 < nk) {
            load_tile((k + 1) << 5);         // overlaps with WMMAs below
            if (k + 2 < nk) {                // L2 hint two tiles ahead
                const int row = tid >> 2, cc = tid & 3;
                __builtin_prefetch(A + (size_t)(bm + row) * K + ((k + 2) << 5) + cc * 8, 0, 1);
                __builtin_prefetch(W + (size_t)(bn + row) * K + ((k + 2) << 5) + cc * 8, 0, 1);
            }
        }
        const int s = k & 1;

        // per-lane WMMA fragments: two ds_load_b128 per 16x32 tile
        union Frag { v16bf v; u32x4 u[2]; };
        Frag a[2], b[4];
#pragma unroll
        for (int mt = 0; mt < 2; ++mt) {
            const bf16* p = &ldsA[s][(wm + mt * 16 + o) * 40];
            a[mt].u[0] = *(const u32x4*)(p + 8 * g);        // K in [8g, 8g+8)
            a[mt].u[1] = *(const u32x4*)(p + 16 + 8 * g);   // K in [16+8g, 24+8g)
        }
#pragma unroll
        for (int nt = 0; nt < 4; ++nt) {
            const bf16* p = &ldsW[s][(wn + nt * 16 + o) * 40];
            b[nt].u[0] = *(const u32x4*)(p + 8 * g);
            b[nt].u[1] = *(const u32x4*)(p + 16 + 8 * g);
        }
#pragma unroll
        for (int mt = 0; mt < 2; ++mt)
#pragma unroll
            for (int nt = 0; nt < 4; ++nt)
                acc[mt][nt] = __builtin_amdgcn_wmma_f32_16x16x32_bf16(
                    false, a[mt].v, false, b[nt].v, (short)0, acc[mt][nt],
                    false, false);

        if (k + 1 < nk) store_tile((k + 1) & 1);
        __syncthreads();
    }

    // epilogue: D layout -> elem r of lane (g,o) is (M = r + 8g, N = o)
#pragma unroll
    for (int mt = 0; mt < 2; ++mt) {
#pragma unroll
        for (int nt = 0; nt < 4; ++nt) {
            const int col = bn + wn + nt * 16 + o;
            const float bc = bias[col];
#pragma unroll
            for (int r = 0; r < 8; ++r) {
                const int row = bm + wm + mt * 16 + 8 * g + r;
                float v = acc[mt][nt][r] + bc;
                if (RESID) v += resid[(size_t)row * N + col];
                if (EPI == 1) v = 0.5f * v * (1.0f + erff(v * 0.70710678118f));
                if (EPI == 2) v = fmaxf(v, 0.f);
                if (OUTF) outF[(size_t)row * N + col] = v;
                if (OUTB) outB[(size_t)row * N + col] = (bf16)v;
            }
        }
    }
}

// ------------------------------------------------------ token+pos embedding
__global__ void __launch_bounds__(256)
embed_kernel(const int* __restrict__ targets, const float* __restrict__ tok_emb,
             const float* __restrict__ pos_emb, float* __restrict__ x)
{
    const size_t i  = (size_t)blockIdx.x * 256 + threadIdx.x;  // over M_DEC*D
    const int    d  = (int)(i & (D_SZ - 1));
    const size_t bt = i >> 10;
    const int    t  = (int)(bt & 3);
    const size_t b  = bt >> 2;
    const int tok = (t == 0) ? NC_SZ : targets[b * NT_SZ + (t - 1)];
    x[i] = tok_emb[(size_t)tok * D_SZ + d] + pos_emb[t * D_SZ + d];
}

// ------------------------------------- causal self-attn, NT=4, dh=128, per (b,h)
// one wave per (b,h); lane owns 4 contiguous channels of dh=128.
__global__ void __launch_bounds__(256)
attn_kernel(const bf16* __restrict__ Q, const bf16* __restrict__ K,
            const bf16* __restrict__ V, bf16* __restrict__ O)
{
    const int idx  = blockIdx.x * 8 + (threadIdx.x >> 5);  // b*H + h
    const int lane = threadIdx.x & 31;
    const int b = idx >> 3;
    const int h = idx & 7;
    const size_t base = (size_t)b * NT_SZ * D_SZ + h * 128 + lane * 4;

    float qf[4][4], kf[4][4], vf[4][4];
#pragma unroll
    for (int t = 0; t < 4; ++t) {
        bf16x4 qv = *(const bf16x4*)(Q + base + t * D_SZ);
        bf16x4 kv = *(const bf16x4*)(K + base + t * D_SZ);
        bf16x4 vv = *(const bf16x4*)(V + base + t * D_SZ);
#pragma unroll
        for (int d = 0; d < 4; ++d) {
            qf[t][d] = (float)qv[d]; kf[t][d] = (float)kv[d]; vf[t][d] = (float)vv[d];
        }
    }
    float s[4][4];
#pragma unroll
    for (int i = 0; i < 4; ++i)
#pragma unroll
        for (int j = 0; j < 4; ++j) {
            float p = 0.f;
#pragma unroll
            for (int d = 0; d < 4; ++d) p += qf[i][d] * kf[j][d];
#pragma unroll
            for (int off = 16; off > 0; off >>= 1) p += __shfl_xor(p, off, 32);
            s[i][j] = p;
        }
    const float sc = 0.088388347648318447f;  // 1/sqrt(128)
    float of[4][4];
#pragma unroll
    for (int i = 0; i < 4; ++i) {
        float mx = -3.4e38f;
#pragma unroll
        for (int j = 0; j < 4; ++j) if (j <= i) mx = fmaxf(mx, s[i][j] * sc);
        float w[4]; float den = 0.f;
#pragma unroll
        for (int j = 0; j < 4; ++j) {
            w[j] = (j <= i) ? __expf(s[i][j] * sc - mx) : 0.f;
            den += w[j];
        }
        const float rden = 1.f / den;
#pragma unroll
        for (int d = 0; d < 4; ++d) {
            float a = 0.f;
#pragma unroll
            for (int j = 0; j < 4; ++j) a += w[j] * vf[j][d];
            of[i][d] = a * rden;
        }
    }
#pragma unroll
    for (int t = 0; t < 4; ++t) {
        bf16x4 ov;
#pragma unroll
        for (int d = 0; d < 4; ++d) ov[d] = (bf16)of[t][d];
        *(bf16x4*)(O + base + t * D_SZ) = ov;
    }
}

// --------------------------- x[b,t,:] += y[b,:]  (degenerate cross-attention)
__global__ void __launch_bounds__(256)
bcast_add_kernel(float* __restrict__ x, const float* __restrict__ y)
{
    const size_t i = (size_t)blockIdx.x * 256 + threadIdx.x;  // over M_DEC*D
    const int    d = (int)(i & (D_SZ - 1));
    const size_t b = i >> 12;                                 // NT*D = 4096
    x[i] += y[b * D_SZ + d];
}

// ----------------------------------------------- logits: 16 classes per row
__global__ void __launch_bounds__(256)
logits_kernel(const bf16* __restrict__ X, const bf16* __restrict__ Wc,
              const float* __restrict__ bc, float* __restrict__ out)
{
    __shared__ bf16 w[NC_SZ * D_SZ];   // 32 KB: whole out_w in LDS
    for (int i = threadIdx.x; i < NC_SZ * D_SZ / 8; i += 256)
        ((u32x4*)w)[i] = ((const u32x4*)Wc)[i];
    __syncthreads();
    const int row  = blockIdx.x * 8 + (threadIdx.x >> 5);
    const int lane = threadIdx.x & 31;
    const bf16* xr = X + (size_t)row * D_SZ;
    float acc[NC_SZ];
#pragma unroll
    for (int c = 0; c < NC_SZ; ++c) acc[c] = 0.f;
#pragma unroll 2
    for (int i = 0; i < 8; ++i) {                 // 8 groups of 4 channels
        const int grp = lane + 32 * i;
        bf16x4 xv = ((const bf16x4*)xr)[grp];
#pragma unroll
        for (int c = 0; c < NC_SZ; ++c) {
            bf16x4 wv = ((const bf16x4*)(w + c * D_SZ))[grp];
            acc[c] += (float)xv[0] * (float)wv[0] + (float)xv[1] * (float)wv[1]
                    + (float)xv[2] * (float)wv[2] + (float)xv[3] * (float)wv[3];
        }
    }
#pragma unroll
    for (int c = 0; c < NC_SZ; ++c) {
        float r = acc[c];
#pragma unroll
        for (int off = 16; off > 0; off >>= 1) r += __shfl_xor(r, off, 32);
        if (lane == c) out[(size_t)row * NC_SZ + c] = r + bc[c];
    }
}

// ============================================================== launch
extern "C" void kernel_launch(void* const* d_in, const int* in_sizes, int n_in,
                              void* d_out, int out_size, void* d_ws, size_t ws_size,
                              hipStream_t stream)
{
    (void)in_sizes; (void)n_in; (void)out_size; (void)ws_size;
    const float* context = (const float*)d_in[0];
    const int*   targets = (const int*)d_in[1];
    const float* tok_emb = (const float*)d_in[2];
    const float* pos_emb = (const float*)d_in[3];
    const float* cp_ln_g = (const float*)d_in[4];
    const float* cp_ln_b = (const float*)d_in[5];
    const float* cp_w    = (const float*)d_in[6];
    const float* cp_b    = (const float*)d_in[7];
    const float* sa_wq   = (const float*)d_in[8];
    const float* sa_bq   = (const float*)d_in[9];
    const float* sa_wk   = (const float*)d_in[10];
    const float* sa_bk   = (const float*)d_in[11];
    const float* sa_wv   = (const float*)d_in[12];
    const float* sa_bv   = (const float*)d_in[13];
    const float* sa_wo   = (const float*)d_in[14];
    const float* sa_bo   = (const float*)d_in[15];
    // d_in[16..19] (ca_wq/bq, ca_wk/bk) and d_in[26..27] (ln2) are exactly
    // unused: softmax over a single cross-attn key is identically 1.
    const float* ca_wv   = (const float*)d_in[20];
    const float* ca_bv   = (const float*)d_in[21];
    const float* ca_wo   = (const float*)d_in[22];
    const float* ca_bo   = (const float*)d_in[23];
    const float* ln1_g   = (const float*)d_in[24];
    const float* ln1_b   = (const float*)d_in[25];
    const float* ln3_g   = (const float*)d_in[28];
    const float* ln3_b   = (const float*)d_in[29];
    const float* ff_w1   = (const float*)d_in[30];
    const float* ff_b1   = (const float*)d_in[31];
    const float* ff_w2   = (const float*)d_in[32];
    const float* ff_b2   = (const float*)d_in[33];
    const float* out_w   = (const float*)d_in[34];
    const float* out_b   = (const float*)d_in[35];
    float* logits = (float*)d_out;

    // ---- workspace carve (~1.5 GB) ----
    char* ws = (char*)d_ws;
    size_t off = 0;
    auto carve = [&](size_t bytes) -> void* {
        void* p = ws + off;
        off += (bytes + 255) & ~(size_t)255;
        return p;
    };
    bf16* cpw_b  = (bf16*)carve((size_t)D_SZ * D_SZ * 2);
    bf16* sawq_b = (bf16*)carve((size_t)D_SZ * D_SZ * 2);
    bf16* sawk_b = (bf16*)carve((size_t)D_SZ * D_SZ * 2);
    bf16* sawv_b = (bf16*)carve((size_t)D_SZ * D_SZ * 2);
    bf16* sawo_b = (bf16*)carve((size_t)D_SZ * D_SZ * 2);
    bf16* cawv_b = (bf16*)carve((size_t)D_SZ * D_SZ * 2);
    bf16* cawo_b = (bf16*)carve((size_t)D_SZ * D_SZ * 2);
    bf16* ffw1_b = (bf16*)carve((size_t)DFF_SZ * D_SZ * 2);
    bf16* ffw2_b = (bf16*)carve((size_t)D_SZ * DFF_SZ * 2);
    bf16* outw_b = (bf16*)carve((size_t)NC_SZ * D_SZ * 2);
    bf16*  a0    = (bf16*)carve((size_t)B_SZ * D_SZ * 2);
    bf16*  memb  = (bf16*)carve((size_t)B_SZ * D_SZ * 2);
    bf16*  vmemb = (bf16*)carve((size_t)B_SZ * D_SZ * 2);
    float* ybuf  = (float*)carve((size_t)B_SZ * D_SZ * 4);
    float* xbuf  = (float*)carve((size_t)M_DEC * D_SZ * 4);
    bf16*  hbuf  = (bf16*)carve((size_t)M_DEC * D_SZ * 2);
    bf16*  qb    = (bf16*)carve((size_t)M_DEC * D_SZ * 2);
    bf16*  kb    = (bf16*)carve((size_t)M_DEC * D_SZ * 2);
    bf16*  vb    = (bf16*)carve((size_t)M_DEC * D_SZ * 2);
    bf16*  attnb = (bf16*)carve((size_t)M_DEC * D_SZ * 2);
    bf16*  f1b   = (bf16*)carve((size_t)M_DEC * DFF_SZ * 2);
    bf16*  xfb   = (bf16*)carve((size_t)M_DEC * D_SZ * 2);

    const dim3 blk(256);
    auto cvt = [&](const float* src, bf16* dst, int n) {
        cvt_bf16_kernel<<<(n + 255) / 256, blk, 0, stream>>>(src, dst, n);
    };
    cvt(cp_w,  cpw_b,  D_SZ * D_SZ);
    cvt(sa_wq, sawq_b, D_SZ * D_SZ);
    cvt(sa_wk, sawk_b, D_SZ * D_SZ);
    cvt(sa_wv, sawv_b, D_SZ * D_SZ);
    cvt(sa_wo, sawo_b, D_SZ * D_SZ);
    cvt(ca_wv, cawv_b, D_SZ * D_SZ);
    cvt(ca_wo, cawo_b, D_SZ * D_SZ);
    cvt(ff_w1, ffw1_b, DFF_SZ * D_SZ);
    cvt(ff_w2, ffw2_b, D_SZ * DFF_SZ);
    cvt(out_w, outw_b, NC_SZ * D_SZ);

    // 1) context path: mem = gelu(LN(context) @ cp_w^T + cp_b)
    ln_bf16_kernel<<<B_SZ / 8, blk, 0, stream>>>(context, cp_ln_g, cp_ln_b, a0);
    gemm_bf16_wmma<1, false, false, true>
        <<<dim3(B_SZ / 128, D_SZ / 128), blk, 0, stream>>>(
            a0, cpw_b, cp_b, nullptr, nullptr, memb, B_SZ, D_SZ, D_SZ);
    // degenerate cross-attn: y = (mem @ ca_wv^T + b_v) @ ca_wo^T + b_o
    gemm_bf16_wmma<0, false, false, true>
        <<<dim3(B_SZ / 128, D_SZ / 128), blk, 0, stream>>>(
            memb, cawv_b, ca_bv, nullptr, nullptr, vmemb, B_SZ, D_SZ, D_SZ);
    gemm_bf16_wmma<0, false, true, false>
        <<<dim3(B_SZ / 128, D_SZ / 128), blk, 0, stream>>>(
            vmemb, cawo_b, ca_bo, nullptr, ybuf, nullptr, B_SZ, D_SZ, D_SZ);

    // 2) decoder embeddings + self-attention block
    embed_kernel<<<(M_DEC * D_SZ) / 256, blk, 0, stream>>>(targets, tok_emb,
                                                           pos_emb, xbuf);
    ln_bf16_kernel<<<M_DEC / 8, blk, 0, stream>>>(xbuf, ln1_g, ln1_b, hbuf);
    gemm_bf16_wmma<0, false, false, true>
        <<<dim3(M_DEC / 128, D_SZ / 128), blk, 0, stream>>>(
            hbuf, sawq_b, sa_bq, nullptr, nullptr, qb, M_DEC, D_SZ, D_SZ);
    gemm_bf16_wmma<0, false, false, true>
        <<<dim3(M_DEC / 128, D_SZ / 128), blk, 0, stream>>>(
            hbuf, sawk_b, sa_bk, nullptr, nullptr, kb, M_DEC, D_SZ, D_SZ);
    gemm_bf16_wmma<0, false, false, true>
        <<<dim3(M_DEC / 128, D_SZ / 128), blk, 0, stream>>>(
            hbuf, sawv_b, sa_bv, nullptr, nullptr, vb, M_DEC, D_SZ, D_SZ);
    attn_kernel<<<(B_SZ * H_SZ) / 8, blk, 0, stream>>>(qb, kb, vb, attnb);
    gemm_bf16_wmma<0, true, true, false>   // x += attn @ sa_wo^T + b (in place)
        <<<dim3(M_DEC / 128, D_SZ / 128), blk, 0, stream>>>(
            attnb, sawo_b, sa_bo, xbuf, xbuf, nullptr, M_DEC, D_SZ, D_SZ);

    // 3) add broadcast cross-attn output
    bcast_add_kernel<<<(M_DEC * D_SZ) / 256, blk, 0, stream>>>(xbuf, ybuf);

    // 4) FFN block
    ln_bf16_kernel<<<M_DEC / 8, blk, 0, stream>>>(xbuf, ln3_g, ln3_b, hbuf);
    gemm_bf16_wmma<2, false, false, true>
        <<<dim3(M_DEC / 128, DFF_SZ / 128), blk, 0, stream>>>(
            hbuf, ffw1_b, ff_b1, nullptr, nullptr, f1b, M_DEC, DFF_SZ, D_SZ);
    gemm_bf16_wmma<0, true, false, true>   // xf = x + f1 @ ff_w2^T + b (bf16)
        <<<dim3(M_DEC / 128, D_SZ / 128), blk, 0, stream>>>(
            f1b, ffw2_b, ff_b2, xbuf, nullptr, xfb, M_DEC, D_SZ, DFF_SZ);

    // 5) logits
    logits_kernel<<<M_DEC / 8, blk, 0, stream>>>(xfb, outw_b, out_b, logits);
}
